// GatingAttentionOpti_24292335027056
// MI455X (gfx1250) — matrix operand order repaired and checked by reading, not compiled, not run on hardware
//
#include <hip/hip_runtime.h>

typedef __attribute__((ext_vector_type(16))) __bf16 v16bf;
typedef __attribute__((ext_vector_type(8)))  __bf16 v8bf;
typedef __attribute__((ext_vector_type(8)))  float  v8f;

#define B_   64
#define H_   8
#define NQ_  512
#define NK_  512
#define C_   32
#define AD_  256

// 16-bit A-matrix 16x32 fragment: element e -> K index (per ISA 7.12.2)
__device__ __forceinline__ int kmapA(int e, int half) {
  return (e & 7) + 8 * half + ((e & 8) << 1);
}
// 16-bit B-matrix 32x16 fragment: element e -> K index
__device__ __forceinline__ int kmapB(int e, int half) {
  return (e & 7) + 8 * half + ((e & 8) << 1);
}

__device__ __forceinline__ v8f wmma_bf16(v16bf a, v16bf b, v8f c) {
  return __builtin_amdgcn_wmma_f32_16x16x32_bf16(false, a, false, b, (short)0, c,
                                                 false, false);
}

// ---------------------------------------------------------------------------
// Kernel 0a: weight prep. Transpose + convert 5 weight matrices [256,256] to
// bf16 [out_col][a] so B-fragment loads are contiguous per lane.
// ---------------------------------------------------------------------------
__global__ __launch_bounds__(256)
void wprep_kernel(const float* __restrict__ qw, const float* __restrict__ kw,
                  const float* __restrict__ vw, const float* __restrict__ gw,
                  const float* __restrict__ ow, __bf16* __restrict__ wt) {
  const int w = blockIdx.y;
  const float* src = (w == 0) ? qw : (w == 1) ? kw : (w == 2) ? vw
                   : (w == 3) ? gw : ow;
  __bf16* dst = wt + (size_t)w * AD_ * AD_;
  const int r = blockIdx.x;
  const int c = threadIdx.x;
  dst[(size_t)c * AD_ + r] = (__bf16)src[(size_t)r * AD_ + c];
}

// ---------------------------------------------------------------------------
// Kernel 0b: activation prep. Convert q_data / m_data fp32 -> bf16 once,
// so proj A-fragments load bf16 directly (no per-wave redundant cvt).
// ---------------------------------------------------------------------------
__global__ __launch_bounds__(256)
void xprep_kernel(const float* __restrict__ qd, const float* __restrict__ md,
                  __bf16* __restrict__ xb) {
  const size_t N = (size_t)B_ * NQ_ * AD_;
  const float* src = blockIdx.y ? md : qd;
  __bf16* dst = xb + (size_t)blockIdx.y * N;
  const size_t i = ((size_t)blockIdx.x * 256 + threadIdx.x) * 8;
  const float4 a = *(const float4*)(src + i);
  const float4 b = *(const float4*)(src + i + 4);
  v8bf o;
  o[0] = (__bf16)a.x; o[1] = (__bf16)a.y; o[2] = (__bf16)a.z; o[3] = (__bf16)a.w;
  o[4] = (__bf16)b.x; o[5] = (__bf16)b.y; o[6] = (__bf16)b.z; o[7] = (__bf16)b.w;
  *(v8bf*)(dst + i) = o;
}

// ---------------------------------------------------------------------------
// Kernel 1: fused projections, 16x128 output per wave (8 accumulators).
// z = 0:q(scaled) 1:k 2:v 3:gate(sigmoid)
// ---------------------------------------------------------------------------
__global__ __launch_bounds__(32)
void proj_kernel(const __bf16* __restrict__ xb, const __bf16* __restrict__ wts,
                 const float* __restrict__ gb,
                 __bf16* __restrict__ qbuf, __bf16* __restrict__ kbuf,
                 __bf16* __restrict__ vbuf, float* __restrict__ gatebuf) {
  const int lane = threadIdx.x & 31;
  const int half = lane >> 4, l16 = lane & 15;
  const int tm = blockIdx.x & 31;   // 32 row tiles of 16
  const int cg = blockIdx.x >> 5;   // 2 col groups of 128
  const int b  = blockIdx.y;
  const int w  = blockIdx.z;

  const size_t N = (size_t)B_ * NQ_ * AD_;
  const __bf16* X  = xb + ((w == 1 || w == 2) ? N : 0);
  const __bf16* WT = wts + (size_t)w * AD_ * AD_;  // [col][a] bf16

  const __bf16* Xrow = X + ((size_t)b * NQ_ + tm * 16 + l16) * AD_;
  v8f acc[8] = {};

  #pragma unroll
  for (int kk = 0; kk < AD_ / 32; ++kk) {
    const int k0 = kk * 32;
    v16bf a;
    #pragma unroll
    for (int e = 0; e < 16; ++e) a[e] = Xrow[k0 + kmapA(e, half)];
    #pragma unroll
    for (int t = 0; t < 8; ++t) {
      const __bf16* Wt = WT + (size_t)(cg * 128 + t * 16 + l16) * AD_;
      v16bf bf;
      #pragma unroll
      for (int e = 0; e < 16; ++e) bf[e] = Wt[k0 + kmapB(e, half)];
      acc[t] = wmma_bf16(a, bf, acc[t]);
    }
  }

  const float scale = 0.17677669529663687f;  // 1/sqrt(32)
  #pragma unroll
  for (int t = 0; t < 8; ++t) {
    const int col = cg * 128 + t * 16 + l16;
    const int h = col >> 5, c = col & 31;
    #pragma unroll
    for (int r = 0; r < 8; ++r) {
      const int m = tm * 16 + r + 8 * half;
      const float v = acc[t][r];
      if (w == 0) {
        qbuf[(((size_t)b * H_ + h) * NQ_ + m) * C_ + c] = (__bf16)(v * scale);
      } else if (w == 1) {
        kbuf[(((size_t)b * H_ + h) * NK_ + m) * C_ + c] = (__bf16)v;
      } else if (w == 2) {
        vbuf[(((size_t)b * H_ + h) * NK_ + m) * C_ + c] = (__bf16)v;
      } else {
        const float g = v + gb[col];
        gatebuf[((size_t)b * NQ_ + m) * AD_ + col] = 1.0f / (1.0f + __expf(-g));
      }
    }
  }
}

// ---------------------------------------------------------------------------
// Kernel 2: attention with online softmax. 4 waves/block, 2 q-tiles per wave
// (K/V fragments amortized over 32 q-rows). S = Q*K^T via WMMA, P relaid
// C->A layout through LDS, O += P*V via WMMA. Row-sum reduction deferred to
// the epilogue. Gate fused into the bf16 store.
// ---------------------------------------------------------------------------
__global__ __launch_bounds__(128)
void attn_kernel(const __bf16* __restrict__ qbuf, const __bf16* __restrict__ kbuf,
                 const __bf16* __restrict__ vbuf, const float* __restrict__ bias,
                 const float* __restrict__ nbias, const float* __restrict__ gatebuf,
                 __bf16* __restrict__ wbuf) {
  __shared__ __bf16 plds[8 * 16 * 32];  // 4 waves x 2 P tiles
  const int lane = threadIdx.x & 31;
  const int wave = threadIdx.x >> 5;
  const int half = lane >> 4, l16 = lane & 15;
  const int qt0 = (blockIdx.x * 4 + wave) * 2;  // 2 of 32 q-tiles
  const int h  = blockIdx.y;
  const int b  = blockIdx.z;

  const size_t bh = (size_t)b * H_ + h;
  const __bf16* Q = qbuf + bh * NQ_ * C_;
  const __bf16* K = kbuf + bh * NK_ * C_;
  const __bf16* V = vbuf + bh * NK_ * C_;
  const float* Bb = bias + (size_t)b * NQ_ * NK_;

  const v8f vzero = {};
  v16bf qa[2];
  #pragma unroll
  for (int qi = 0; qi < 2; ++qi) {
    #pragma unroll
    for (int e = 0; e < 16; ++e)
      qa[qi][e] = Q[((qt0 + qi) * 16 + l16) * C_ + kmapA(e, half)];
  }

  v8f o[2][2] = {};
  float rmax[2][8], rsum[2][8];  // rsum = per-lane partials
  #pragma unroll
  for (int qi = 0; qi < 2; ++qi)
    #pragma unroll
    for (int r = 0; r < 8; ++r) { rmax[qi][r] = -3.0e38f; rsum[qi][r] = 0.0f; }

  __bf16* pw[2];
  pw[0] = plds + wave * (2 * 16 * 32);
  pw[1] = pw[0] + 16 * 32;

  for (int ck = 0; ck < NK_ / 32; ++ck) {
    const int nk0 = ck * 32;

    v16bf kb0, kb1;
    #pragma unroll
    for (int e = 0; e < 16; ++e) {
      const int kc = kmapB(e, half);
      kb0[e] = K[(nk0 + l16) * C_ + kc];
      kb1[e] = K[(nk0 + 16 + l16) * C_ + kc];
    }

    v8f s[2][2];
    #pragma unroll
    for (int qi = 0; qi < 2; ++qi) {
      s[qi][0] = wmma_bf16(qa[qi], kb0, vzero);
      s[qi][1] = wmma_bf16(qa[qi], kb1, vzero);
    }

    #pragma unroll
    for (int qi = 0; qi < 2; ++qi) {
      #pragma unroll
      for (int r = 0; r < 8; ++r) {
        const int m = (qt0 + qi) * 16 + r + 8 * half;
        const size_t rowoff = (size_t)m * NK_ + nk0 + l16;
        s[qi][0][r] += Bb[rowoff] + nbias[rowoff];
        s[qi][1][r] += Bb[rowoff + 16] + nbias[rowoff + 16];
      }
      // online softmax; rows live in 16-lane halves -> shfl_xor 1/2/4/8
      #pragma unroll
      for (int r = 0; r < 8; ++r) {
        float mx = fmaxf(s[qi][0][r], s[qi][1][r]);
        mx = fmaxf(mx, __shfl_xor(mx, 1, 32));
        mx = fmaxf(mx, __shfl_xor(mx, 2, 32));
        mx = fmaxf(mx, __shfl_xor(mx, 4, 32));
        mx = fmaxf(mx, __shfl_xor(mx, 8, 32));
        const float mnew = fmaxf(rmax[qi][r], mx);
        const float corr = __expf(rmax[qi][r] - mnew);  // uniform across half
        rmax[qi][r] = mnew;
        const float p0 = __expf(s[qi][0][r] - mnew);
        const float p1 = __expf(s[qi][1][r] - mnew);
        rsum[qi][r] = rsum[qi][r] * corr + (p0 + p1);
        o[qi][0][r] *= corr;
        o[qi][1][r] *= corr;
        pw[qi][(r + 8 * half) * 32 + l16]      = (__bf16)p0;
        pw[qi][(r + 8 * half) * 32 + 16 + l16] = (__bf16)p1;
      }
    }
    asm volatile("s_wait_dscnt 0" ::: "memory");

    v16bf pa[2];
    #pragma unroll
    for (int qi = 0; qi < 2; ++qi)
      #pragma unroll
      for (int e = 0; e < 16; ++e) pa[qi][e] = pw[qi][l16 * 32 + kmapA(e, half)];

    v16bf vb0, vb1;
    #pragma unroll
    for (int e = 0; e < 16; ++e) {
      const int kr = nk0 + kmapB(e, half);
      vb0[e] = V[kr * C_ + l16];
      vb1[e] = V[kr * C_ + 16 + l16];
    }
    #pragma unroll
    for (int qi = 0; qi < 2; ++qi) {
      o[qi][0] = wmma_bf16(pa[qi], vb0, o[qi][0]);
      o[qi][1] = wmma_bf16(pa[qi], vb1, o[qi][1]);
    }
  }

  #pragma unroll
  for (int qi = 0; qi < 2; ++qi) {
    #pragma unroll
    for (int r = 0; r < 8; ++r) {
      float s = rsum[qi][r];  // final cross-lane row-sum reduction
      s += __shfl_xor(s, 1, 32);
      s += __shfl_xor(s, 2, 32);
      s += __shfl_xor(s, 4, 32);
      s += __shfl_xor(s, 8, 32);
      const float inv = 1.0f / s;
      const int m = (qt0 + qi) * 16 + r + 8 * half;
      const size_t i0 = ((size_t)b * NQ_ + m) * AD_ + h * C_ + l16;
      wbuf[i0]      = (__bf16)(o[qi][0][r] * inv * gatebuf[i0]);
      wbuf[i0 + 16] = (__bf16)(o[qi][1][r] * inv * gatebuf[i0 + 16]);
    }
  }
}

// ---------------------------------------------------------------------------
// Kernel 3: output projection, 16x128 per wave.
// out = wbuf(bf16)[32768,256] x output_w[256,256] + b  (fp32 out)
// ---------------------------------------------------------------------------
__global__ __launch_bounds__(32)
void out_kernel(const __bf16* __restrict__ wbuf, const __bf16* __restrict__ owt,
                const float* __restrict__ ob, float* __restrict__ out) {
  const int lane = threadIdx.x & 31;
  const int half = lane >> 4, l16 = lane & 15;
  const int tm = blockIdx.x & 31;
  const int cg = blockIdx.x >> 5;
  const int b  = blockIdx.y;

  const __bf16* Xrow = wbuf + ((size_t)b * NQ_ + tm * 16 + l16) * AD_;
  v8f acc[8] = {};

  #pragma unroll
  for (int kk = 0; kk < AD_ / 32; ++kk) {
    const int k0 = kk * 32;
    v16bf a;
    #pragma unroll
    for (int e = 0; e < 16; ++e) a[e] = Xrow[k0 + kmapA(e, half)];
    #pragma unroll
    for (int t = 0; t < 8; ++t) {
      const __bf16* Wt = owt + (size_t)(cg * 128 + t * 16 + l16) * AD_;
      v16bf bf;
      #pragma unroll
      for (int e = 0; e < 16; ++e) bf[e] = Wt[k0 + kmapB(e, half)];
      acc[t] = wmma_bf16(a, bf, acc[t]);
    }
  }

  #pragma unroll
  for (int t = 0; t < 8; ++t) {
    const int col = cg * 128 + t * 16 + l16;
    #pragma unroll
    for (int r = 0; r < 8; ++r) {
      const int m = tm * 16 + r + 8 * half;
      out[((size_t)b * NQ_ + m) * AD_ + col] = acc[t][r] + ob[col];
    }
  }
}

// ---------------------------------------------------------------------------
extern "C" void kernel_launch(void* const* d_in, const int* in_sizes, int n_in,
                              void* d_out, int out_size, void* d_ws, size_t ws_size,
                              hipStream_t stream) {
  const float* qd = (const float*)d_in[0];
  const float* md = (const float*)d_in[1];
  const float* bias = (const float*)d_in[2];
  const float* nb = (const float*)d_in[3];
  const float* qw = (const float*)d_in[4];
  const float* kw = (const float*)d_in[5];
  const float* vw = (const float*)d_in[6];
  const float* gw = (const float*)d_in[7];
  const float* gb = (const float*)d_in[8];
  const float* ow = (const float*)d_in[9];
  const float* ob = (const float*)d_in[10];
  float* out = (float*)d_out;

  // workspace layout (all intermediates L2-resident, ~135MB total):
  // xb    bf16 [2][B,512,256]  33.5MB  @ 0
  // qbuf  bf16 [B,H,512,32]    16.8MB  @ 33554432
  // kbuf  bf16                 16.8MB  @ 50331648
  // vbuf  bf16                 16.8MB  @ 67108864
  // gate  f32  [B,512,256]     33.5MB  @ 83886080
  // wbuf  bf16 [B,512,256]     16.8MB  @ 117440512
  // wt    bf16 5x[256,256]      0.6MB  @ 134217728
  char* ws = (char*)d_ws;
  __bf16* xb     = (__bf16*)(ws);
  __bf16* qbuf   = (__bf16*)(ws + (size_t)33554432);
  __bf16* kbuf   = (__bf16*)(ws + (size_t)50331648);
  __bf16* vbuf   = (__bf16*)(ws + (size_t)67108864);
  float*  gatebuf = (float*)(ws + (size_t)83886080);
  __bf16* wbuf   = (__bf16*)(ws + (size_t)117440512);
  __bf16* wt     = (__bf16*)(ws + (size_t)134217728);
  __bf16* owt    = wt + (size_t)4 * AD_ * AD_;

  dim3 gw0(AD_, 5, 1);
  wprep_kernel<<<gw0, 256, 0, stream>>>(qw, kw, vw, gw, ow, wt);

  dim3 gx0(4096, 2, 1);  // 2 sources, 8 elems/thread
  xprep_kernel<<<gx0, 256, 0, stream>>>(qd, md, xb);

  dim3 g1(64, B_, 4);    // x = tm(32) * colgroup(2)
  proj_kernel<<<g1, 32, 0, stream>>>(xb, wt, gb, qbuf, kbuf, vbuf, gatebuf);

  dim3 g2(4, H_, B_);    // 4 waves/block, 2 q-tiles/wave -> 32 q-tiles
  attn_kernel<<<g2, 128, 0, stream>>>(qbuf, kbuf, vbuf, bias, nb, gatebuf, wbuf);

  dim3 g3(64, B_, 1);
  out_kernel<<<g3, 32, 0, stream>>>(wbuf, owt, ob, out);
}
